// HyperConnection_36275293782720
// MI455X (gfx1250) — matrix operand order repaired and checked by reading, not compiled, not run on hardware
//
#include <hip/hip_runtime.h>
#include <hip/hip_bf16.h>

// ---------------------------------------------------------------------------
// HyperConnection forward for MI455X (gfx1250, wave32, WMMA).
//   N=4 streams, D=896, nD=3584, tokens = B*S = 16384.
//   Projection GEMM (16384 x 3584) x (3584 x 24) done with
//   v_wmma_f32_16x16x32_bf16; Sinkhorn + output mixing in VALU.
// ---------------------------------------------------------------------------

#define NN   4
#define DD   896
#define ND   3584          // NN*DD
#define NTOK 16384         // B*S
#define KS   112           // ND / 32 k-slices
#define WAVES 4            // waves per workgroup
#define EPSV 1e-6f

typedef __attribute__((ext_vector_type(16))) __bf16 v16bf;
typedef __attribute__((ext_vector_type(8)))  float  v8f;
typedef __attribute__((ext_vector_type(4)))  float  f4;

__device__ __forceinline__ float sigmoidf_(float t) {
    return 1.0f / (1.0f + __expf(-t));
}

// ---------------------------------------------------------------------------
// Prep: pack [theta_pre | theta_post | theta_res] transposed to bf16.
// theta_t[row][k], row-pitch ND.  rows 0-3: pre, 4-7: post, 8-23: res, 24-31: 0
// ---------------------------------------------------------------------------
__global__ void hc_prep_theta(const float* __restrict__ tpre,
                              const float* __restrict__ tpost,
                              const float* __restrict__ tres,
                              __bf16* __restrict__ theta_t) {
    int idx = blockIdx.x * blockDim.x + threadIdx.x;   // 32*ND total
    if (idx >= 32 * ND) return;
    int row = idx / ND;
    int k   = idx - row * ND;
    float v = 0.0f;
    if (row < 4)        v = tpre [k * 4  + row];
    else if (row < 8)   v = tpost[k * 4  + (row - 4)];
    else if (row < 24)  v = tres [k * 16 + (row - 8)];
    theta_t[(size_t)row * ND + k] = (__bf16)v;
}

// ---------------------------------------------------------------------------
// Main kernel: one wave per 16-token tile.
// ---------------------------------------------------------------------------
__global__ void __launch_bounds__(WAVES * 32)
hc_main(const float* __restrict__ x,
        const float* __restrict__ f_out,
        const float* __restrict__ norm_w,
        const float* __restrict__ b_pre,
        const float* __restrict__ alpha_pre,
        const float* __restrict__ b_post,
        const float* __restrict__ alpha_post,
        const float* __restrict__ b_res,
        const float* __restrict__ alpha_res,
        const __bf16* __restrict__ theta_t,
        float* __restrict__ out) {

    __shared__ float s_normw[ND];                 // 14336 B
    __shared__ float s_proj [WAVES][16][32];      //  8192 B
    __shared__ float s_sumsq[WAVES][16];          //   256 B
    __shared__ float s_h    [WAVES][16][24];      //  6144 B

    const int tid   = threadIdx.x;
    const int wave  = tid >> 5;
    const int lane  = tid & 31;
    const int m     = lane & 15;       // A row / B column / C column index
    const int khalf = lane >> 4;       // which K half this lane covers

    // stage norm_w once per workgroup
    for (int i = tid; i < ND; i += blockDim.x) s_normw[i] = norm_w[i];
    __syncthreads();

    const int tile = blockIdx.x * WAVES + wave;   // 16-token tile id
    const int tokA = tile * 16 + m;               // token this lane feeds into A
    const float* xrow = x + (size_t)tokA * ND;

    // ---------------- Phase 1: projection GEMM + sum of squares -----------
    v8f acc0 = {};      // proj outputs 0..15
    v8f acc1 = {};      // proj outputs 16..23 (cols 8..15 hit zero-padded B)
    float ssq = 0.0f;

    for (int s = 0; s < KS; ++s) {
        const int kb = s * 32;
        const int o0 = kb + 16 * khalf;       // A elems 0..7  : K = o0..o0+7
        const int o1 = kb + 16 * khalf + 8;   // A elems 8..15 : K = o1..o1+7

        f4 x0 = *(const f4*)(xrow + o0);
        f4 x1 = *(const f4*)(xrow + o0 + 4);
        f4 x2 = *(const f4*)(xrow + o1);
        f4 x3 = *(const f4*)(xrow + o1 + 4);

        ssq = fmaf(x0.x, x0.x, fmaf(x0.y, x0.y, fmaf(x0.z, x0.z, fmaf(x0.w, x0.w, ssq))));
        ssq = fmaf(x1.x, x1.x, fmaf(x1.y, x1.y, fmaf(x1.z, x1.z, fmaf(x1.w, x1.w, ssq))));
        ssq = fmaf(x2.x, x2.x, fmaf(x2.y, x2.y, fmaf(x2.z, x2.z, fmaf(x2.w, x2.w, ssq))));
        ssq = fmaf(x3.x, x3.x, fmaf(x3.y, x3.y, fmaf(x3.z, x3.z, fmaf(x3.w, x3.w, ssq))));

        f4 w0 = *(const f4*)&s_normw[o0];
        f4 w1 = *(const f4*)&s_normw[o0 + 4];
        f4 w2 = *(const f4*)&s_normw[o1];
        f4 w3 = *(const f4*)&s_normw[o1 + 4];

        v16bf a;
        a[0]  = (__bf16)(x0.x * w0.x); a[1]  = (__bf16)(x0.y * w0.y);
        a[2]  = (__bf16)(x0.z * w0.z); a[3]  = (__bf16)(x0.w * w0.w);
        a[4]  = (__bf16)(x1.x * w1.x); a[5]  = (__bf16)(x1.y * w1.y);
        a[6]  = (__bf16)(x1.z * w1.z); a[7]  = (__bf16)(x1.w * w1.w);
        a[8]  = (__bf16)(x2.x * w2.x); a[9]  = (__bf16)(x2.y * w2.y);
        a[10] = (__bf16)(x2.z * w2.z); a[11] = (__bf16)(x2.w * w2.w);
        a[12] = (__bf16)(x3.x * w3.x); a[13] = (__bf16)(x3.y * w3.y);
        a[14] = (__bf16)(x3.z * w3.z); a[15] = (__bf16)(x3.w * w3.w);

        // B fragments: lane = column m; contiguous 16 K values per lane half
        const int kbB = kb + 16 * khalf;
        v16bf b0 = *(const v16bf*)(theta_t + (size_t)m        * ND + kbB);
        v16bf b1 = *(const v16bf*)(theta_t + (size_t)(16 + m) * ND + kbB);

        acc0 = __builtin_amdgcn_wmma_f32_16x16x32_bf16(false, a, false, b0,
                                                       (short)0, acc0, false, false);
        acc1 = __builtin_amdgcn_wmma_f32_16x16x32_bf16(false, a, false, b1,
                                                       (short)0, acc1, false, false);
    }

    // combine the two K halves of sum-of-squares (lanes m and m+16)
    float ssq_tok = ssq + __shfl_xor(ssq, 16);
    if (lane < 16) s_sumsq[wave][lane] = ssq_tok;

    // scatter C into LDS:  C layout -> token row = r + 8*khalf, proj = m
#pragma unroll
    for (int r = 0; r < 8; ++r) {
        s_proj[wave][r + 8 * khalf][m]      = acc0[r];
        s_proj[wave][r + 8 * khalf][16 + m] = acc1[r];
    }
    __syncthreads();

    // ---------------- Phase 2: routing weights + Sinkhorn -----------------
    if (lane < 16) {
        float raw[24];
#pragma unroll
        for (int p = 0; p < 24; ++p) raw[p] = s_proj[wave][lane][p];
        float scale = rsqrtf(s_sumsq[wave][lane] * (1.0f / ND) + EPSV);

        const float ap  = alpha_pre[0];
        const float apo = alpha_post[0];
        const float ar  = alpha_res[0];

        float h[24];
#pragma unroll
        for (int j = 0; j < 4; ++j) {
            h[j]     = sigmoidf_(ap  * scale * raw[j]     + b_pre[j]);
            h[4 + j] = 2.0f * sigmoidf_(apo * scale * raw[4 + j] + b_post[j]);
        }

        float M[4][4];
#pragma unroll
        for (int i = 0; i < 4; ++i)
#pragma unroll
            for (int j = 0; j < 4; ++j)
                M[i][j] = ar * scale * raw[8 + i * 4 + j] + b_res[i * 4 + j];

        for (int it = 0; it < 20; ++it) {
#pragma unroll
            for (int i = 0; i < 4; ++i) {     // row logsumexp
                float mx = fmaxf(fmaxf(M[i][0], M[i][1]), fmaxf(M[i][2], M[i][3]));
                float sm = __expf(M[i][0] - mx) + __expf(M[i][1] - mx)
                         + __expf(M[i][2] - mx) + __expf(M[i][3] - mx);
                float l  = mx + __logf(sm);
#pragma unroll
                for (int j = 0; j < 4; ++j) M[i][j] -= l;
            }
#pragma unroll
            for (int j = 0; j < 4; ++j) {     // column logsumexp
                float mx = fmaxf(fmaxf(M[0][j], M[1][j]), fmaxf(M[2][j], M[3][j]));
                float sm = __expf(M[0][j] - mx) + __expf(M[1][j] - mx)
                         + __expf(M[2][j] - mx) + __expf(M[3][j] - mx);
                float l  = mx + __logf(sm);
#pragma unroll
                for (int i = 0; i < 4; ++i) M[i][j] -= l;
            }
        }
#pragma unroll
        for (int i = 0; i < 4; ++i)
#pragma unroll
            for (int j = 0; j < 4; ++j)
                h[8 + i * 4 + j] = __expf(M[i][j]);

#pragma unroll
        for (int p = 0; p < 24; ++p) s_h[wave][lane][p] = h[p];
    }
    __syncthreads();

    // ---------------- Phase 3: output mixing (memory bound) ---------------
    const int lane4 = lane * 4;
    for (int t = 0; t < 16; ++t) {
        const int tk = tile * 16 + t;
        float hv[24];
#pragma unroll
        for (int p = 0; p < 24; ++p) hv[p] = s_h[wave][t][p];   // LDS broadcast

        const float* xr   = x     + (size_t)tk * ND;
        const float* fr   = f_out + (size_t)tk * DD;
        float*       orow = out   + (size_t)tk * 5 * DD;

#pragma unroll
        for (int c = 0; c < 7; ++c) {
            const int d = c * 128 + lane4;
            f4 xv0 = *(const f4*)(xr + 0 * DD + d);
            f4 xv1 = *(const f4*)(xr + 1 * DD + d);
            f4 xv2 = *(const f4*)(xr + 2 * DD + d);
            f4 xv3 = *(const f4*)(xr + 3 * DD + d);
            f4 fv  = __builtin_nontemporal_load((const f4*)(fr + d));

            f4 op = hv[0] * xv0 + hv[1] * xv1 + hv[2] * xv2 + hv[3] * xv3;
            __builtin_nontemporal_store(op, (f4*)(orow + d));

#pragma unroll
            for (int i = 0; i < 4; ++i) {
                f4 oi = hv[8 + i * 4 + 0] * xv0 + hv[8 + i * 4 + 1] * xv1
                      + hv[8 + i * 4 + 2] * xv2 + hv[8 + i * 4 + 3] * xv3
                      + hv[4 + i] * fv;
                __builtin_nontemporal_store(oi, (f4*)(orow + (1 + i) * DD + d));
            }
        }
    }
}

// ---------------------------------------------------------------------------
extern "C" void kernel_launch(void* const* d_in, const int* in_sizes, int n_in,
                              void* d_out, int out_size, void* d_ws, size_t ws_size,
                              hipStream_t stream) {
    const float* x          = (const float*)d_in[0];
    const float* f_out      = (const float*)d_in[1];
    const float* norm_w     = (const float*)d_in[2];
    const float* theta_pre  = (const float*)d_in[3];
    const float* b_pre      = (const float*)d_in[4];
    const float* alpha_pre  = (const float*)d_in[5];
    const float* theta_post = (const float*)d_in[6];
    const float* b_post     = (const float*)d_in[7];
    const float* alpha_post = (const float*)d_in[8];
    const float* theta_res  = (const float*)d_in[9];
    const float* b_res      = (const float*)d_in[10];
    const float* alpha_res  = (const float*)d_in[11];
    float* out = (float*)d_out;

    __bf16* theta_t = (__bf16*)d_ws;   // 32 * 3584 * 2 = 229376 bytes

    // pack theta -> bf16 transposed B matrix
    {
        const int total = 32 * ND;
        hc_prep_theta<<<(total + 255) / 256, 256, 0, stream>>>(
            theta_pre, theta_post, theta_res, theta_t);
    }

    // main fused kernel: 256 blocks x 128 threads (4 waves), 64 tokens/block
    {
        dim3 grid(NTOK / (16 * WAVES));
        dim3 block(WAVES * 32);
        hc_main<<<grid, block, 0, stream>>>(x, f_out, norm_w,
                                            b_pre, alpha_pre,
                                            b_post, alpha_post,
                                            b_res, alpha_res,
                                            theta_t, out);
    }
}
